// SentimentAnalyzer_30021821399799
// MI455X (gfx1250) — matrix operand order repaired and checked by reading, not compiled, not run on hardware
//
#include <hip/hip_runtime.h>
#include <hip/hip_bf16.h>
#include <math.h>

// Problem constants (from reference)
#define SS 512
#define BB 256
#define EE 128
#define HH 256

typedef __attribute__((ext_vector_type(16))) __bf16 v16bf;
typedef __attribute__((ext_vector_type(8)))  float  v8f;

// ---- fragment loaders -------------------------------------------------------

// B operand, 32x16 bf16, chunk of K=32 read from an f32 row as two runs of 8:
// lane L<16 holds column n, K=0..7,16..23 of chunk; lane 16+L holds K=8..15,24..31.
// Caller passes pointer to this lane's first f32; it reads p[0..7] and p[16..23].
__device__ __forceinline__ v16bf row_b_frag(const float* __restrict__ p) {
    v16bf b;
#pragma unroll
    for (int i = 0; i < 16; ++i) b[i] = (__bf16)p[i < 8 ? i : i + 8];
    return b;
}

// A operand, 16x32 bf16, from a global f32 row (emb row), chunk c (K=32c..32c+31).
// lanes 0-15: v0-3 = K 32c+0..7, v4-7 = K 32c+16..23
// lanes16-31: v0-3 = K 32c+8..15, v4-7 = K 32c+24..31
__device__ __forceinline__ v16bf emb_a_frag(const float* __restrict__ erow, int c, bool hi) {
    const int k1 = 32 * c + (hi ? 8 : 0);
    v16bf a;
#pragma unroll
    for (int i = 0; i < 8; ++i) {
        a[i]     = (__bf16)erow[k1 + i];
        a[8 + i] = (__bf16)erow[k1 + 16 + i];
    }
    return a;
}

// A operand from LDS bf16 h-row (256 bf16), chunk c. Two aligned 16B reads.
__device__ __forceinline__ v16bf lds_a_frag(const __bf16* __restrict__ hrow, int c, bool hi) {
    const int k1 = 32 * c + (hi ? 8 : 0);
    union { uint4 u[2]; v16bf v; } t;
    t.u[0] = *reinterpret_cast<const uint4*>(hrow + k1);
    t.u[1] = *reinterpret_cast<const uint4*>(hrow + k1 + 16);
    return t.v;
}

// ---- fused embed + input-proj + RNN recurrence ------------------------------
// grid = B/16 blocks, 512 threads (16 waves). Wave w owns output columns
// n0 = 16w .. 16w+15; block owns batch rows b0 .. b0+15. No global sync:
// the whole 512-step recurrence runs inside the block.
__global__ void __launch_bounds__(512, 1)
rnn_fused_kernel(const int* __restrict__ X, const float* __restrict__ emb,
                 const float* __restrict__ W_ih, const float* __restrict__ W_hh,
                 const float* __restrict__ b_ih, const float* __restrict__ b_hh,
                 float* __restrict__ out) {
    __shared__ __bf16 hbuf[2][16][HH];      // double-buffered hidden state, 16 KB
    __shared__ int    xidx[SS * 16];        // this block's X column slice, 32 KB

    const int tid  = threadIdx.x;
    const int wave = tid >> 5;
    const int lane = tid & 31;
    const int b0   = blockIdx.x * 16;       // batch rows of this block
    const int n0   = wave * 16;             // output columns of this wave
    const int r    = lane & 15;             // tile row (A/C) == tile col (B/C)
    const bool hi  = lane >= 16;
    const int n    = n0 + r;                // this lane's output column

    // Stage X indices for all timesteps of our 16 batch rows.
    for (int i = tid; i < SS * 16; i += 512) {
        const int s  = i >> 4;
        const int rr = i & 15;
        xidx[i] = X[s * BB + b0 + rr];
    }
    // h0 = 0
    for (int i = tid; i < 16 * HH; i += 512)
        hbuf[0][i >> 8][i & 255] = (__bf16)0.0f;

    // Resident B fragments: W_hh (8 K-chunks) and W_ih (4 K-chunks).
    // B[k][n] = W[n][k]  -> each lane reads two runs of 8 f32 of row n, 16 apart.
    v16bf Bhh[8];
    {
        const float* base = W_hh + (size_t)n * HH + (hi ? 8 : 0);
#pragma unroll
        for (int c = 0; c < 8; ++c) Bhh[c] = row_b_frag(base + 32 * c);
    }
    v16bf Bih[4];
    {
        const float* base = W_ih + (size_t)n * EE + (hi ? 8 : 0);
#pragma unroll
        for (int c = 0; c < 4; ++c) Bih[c] = row_b_frag(base + 32 * c);
    }
    const float bias = b_ih[n] + b_hh[n];

    __syncthreads();

    // Prefetch emb A-fragments for step 0.
    v16bf Aemb[4];
    {
        const int idx = xidx[r];
        const float* erow = emb + (size_t)idx * EE;
#pragma unroll
        for (int c = 0; c < 4; ++c) Aemb[c] = emb_a_frag(erow, c, hi);
    }

    int cur = 0;
    for (int s = 0; s < SS; ++s) {
        v8f acc;
#pragma unroll
        for (int v = 0; v < 8; ++v) acc[v] = bias;

        // Input projection: xp += embedded @ W_ih^T   (K = E = 128)
#pragma unroll
        for (int c = 0; c < 4; ++c)
            acc = __builtin_amdgcn_wmma_f32_16x16x32_bf16(
                false, Aemb[c], false, Bih[c], (short)0, acc, false, false);

        // Prefetch next step's emb fragments (latency hides under the
        // recurrent WMMAs below; emb table is L2-resident).
        if (s + 1 < SS) {
            const int idx = xidx[(s + 1) * 16 + r];
            const float* erow = emb + (size_t)idx * EE;
#pragma unroll
            for (int c = 0; c < 4; ++c) Aemb[c] = emb_a_frag(erow, c, hi);
        }

        // Recurrent GEMM: acc += h @ W_hh^T   (K = H = 256)
        const __bf16* hrow = &hbuf[cur][r][0];
#pragma unroll
        for (int c = 0; c < 8; ++c) {
            const v16bf a = lds_a_frag(hrow, c, hi);
            acc = __builtin_amdgcn_wmma_f32_16x16x32_bf16(
                false, a, false, Bhh[c], (short)0, acc, false, false);
        }

        // h_new = tanh(acc); write bf16 into the other buffer.
        const int nxt = cur ^ 1;
#pragma unroll
        for (int v = 0; v < 8; ++v) {
            const int m  = v + (hi ? 8 : 0);    // C layout: row m, col n
            const float hv = tanhf(acc[v]);
            hbuf[nxt][m][n] = (__bf16)hv;
            if (s == SS - 1)
                out[(size_t)(b0 + m) * HH + n] = hv;   // f32 h_last -> d_out
        }
        __syncthreads();
        cur = nxt;
    }
}

// ---- in-place log_softmax over rows of d_out --------------------------------
// One wave per row (H = 256 = 32 lanes x 8).
__global__ void __launch_bounds__(256)
logsoftmax_inplace_kernel(float* __restrict__ out) {
    const int row  = blockIdx.x * 8 + (threadIdx.x >> 5);
    const int lane = threadIdx.x & 31;
    float* p = out + (size_t)row * HH;

    float x[8];
    float m = -INFINITY;
#pragma unroll
    for (int j = 0; j < 8; ++j) {
        x[j] = p[lane + 32 * j];
        m = fmaxf(m, x[j]);
    }
#pragma unroll
    for (int off = 16; off > 0; off >>= 1)
        m = fmaxf(m, __shfl_xor(m, off, 32));

    float sum = 0.0f;
#pragma unroll
    for (int j = 0; j < 8; ++j) sum += __expf(x[j] - m);
#pragma unroll
    for (int off = 16; off > 0; off >>= 1)
        sum += __shfl_xor(sum, off, 32);

    const float lse = m + __logf(sum);
#pragma unroll
    for (int j = 0; j < 8; ++j) p[lane + 32 * j] = x[j] - lse;
}

// ---- launch -----------------------------------------------------------------
extern "C" void kernel_launch(void* const* d_in, const int* in_sizes, int n_in,
                              void* d_out, int out_size, void* d_ws, size_t ws_size,
                              hipStream_t stream) {
    (void)in_sizes; (void)n_in; (void)out_size; (void)d_ws; (void)ws_size;
    const int*   X    = (const int*)  d_in[0];
    const float* emb  = (const float*)d_in[1];
    const float* W_ih = (const float*)d_in[2];
    const float* W_hh = (const float*)d_in[3];
    const float* b_ih = (const float*)d_in[4];
    const float* b_hh = (const float*)d_in[5];
    float* out = (float*)d_out;

    rnn_fused_kernel<<<BB / 16, 512, 0, stream>>>(X, emb, W_ih, W_hh, b_ih, b_hh, out);
    logsoftmax_inplace_kernel<<<BB / 8, 256, 0, stream>>>(out);
}
